// BahdanauAttention_10625749090625
// MI455X (gfx1250) — matrix operand order repaired and checked by reading, not compiled, not run on hardware
//
#include <hip/hip_runtime.h>
#include <hip/hip_bf16.h>

typedef __attribute__((ext_vector_type(16))) __bf16 v16bf;
typedef __attribute__((ext_vector_type(8)))  __bf16 v8bf;
typedef __attribute__((ext_vector_type(4)))  __bf16 v4bf;
typedef __attribute__((ext_vector_type(8)))  float  v8f;
typedef __attribute__((ext_vector_type(4)))  float  v4f;
typedef __attribute__((ext_vector_type(4)))  unsigned v4u;
typedef __attribute__((ext_vector_type(8)))  int    v8i;
typedef __attribute__((ext_vector_type(4)))  int    v4i;

#define H_DIM 256
#define S_LEN 4096
#define B_N   32
#define TS    64          // s-rows per score block
#define LSTR  264         // bf16 LDS row stride (528B: 16B aligned, bank-spread)

__device__ __forceinline__ __bf16 f2bf(float f) {
  unsigned u = __builtin_bit_cast(unsigned, f);
  unsigned r = (u + 0x7FFFu + ((u >> 16) & 1u)) >> 16;   // round-to-nearest-even
  return __builtin_bit_cast(__bf16, (unsigned short)r);
}

// ---- Phase 0a: qb[b,o] = query[b]·Wa_w[o,:] + Wa_b[o] + Ua_b[o] ----
__global__ void __launch_bounds__(256)
qproj_kernel(const float* __restrict__ query, const float* __restrict__ Wa_w,
             const float* __restrict__ Wa_b, const float* __restrict__ Ua_b,
             float* __restrict__ qb) {
  int b = blockIdx.x, o = threadIdx.x;
  __shared__ float qs[H_DIM];
  qs[o] = query[b * H_DIM + o];
  __syncthreads();
  float acc = Wa_b[o] + Ua_b[o];
  const float* wrow = Wa_w + (size_t)o * H_DIM;
#pragma unroll 8
  for (int h = 0; h < H_DIM; ++h) acc += qs[h] * wrow[h];
  qb[b * H_DIM + o] = acc;
}

// ---- Phase 0b: Ua_w fp32 -> bf16 (row-major [o][h], B-fragment friendly) ----
__global__ void __launch_bounds__(256)
cvt_ua_kernel(const float* __restrict__ Ua_w, __bf16* __restrict__ ua) {
  int i = blockIdx.x * 256 + threadIdx.x;
  ua[i] = f2bf(Ua_w[i]);
}

// ---- Phase A: scores[b,s] = Va·tanh(keys[b,s]·Ua^T + qb[b]) + Va_b ----
__global__ void __launch_bounds__(256)
score_kernel(const float* __restrict__ keys, const __bf16* __restrict__ ua,
             const float* __restrict__ qb, const float* __restrict__ Va_w,
             const float* __restrict__ Va_b, float* __restrict__ scores) {
  __shared__ float  tile32[TS * H_DIM];   // raw fp32 keys tile (TDM destination)
  __shared__ __bf16 tile[TS * LSTR];      // bf16 tile consumed by WMMA A-frags
  __shared__ float  sc[TS];
  const int b  = blockIdx.y;
  const int s0 = blockIdx.x * TS;
  const int tid = threadIdx.x;
  if (tid < TS) sc[tid] = 0.f;

  // ---- TDM: one tensor_load_to_lds pulls the whole 64x256 fp32 tile ----
  const float* kbase = keys + ((size_t)b * S_LEN + s0) * H_DIM;
  if (tid < 32) {
    unsigned lds_addr = (unsigned)(size_t)(&tile32[0]);
    unsigned long long ga = (unsigned long long)(size_t)kbase;
    v4u g0;
    g0[0] = 1u;                                          // count=1 (user D#)
    g0[1] = lds_addr;                                    // lds_addr[31:0]
    g0[2] = (unsigned)ga;                                // global_addr[31:0]
    g0[3] = (unsigned)((ga >> 32) & 0x01FFFFFFull)       // global_addr[56:32]
          | (2u << 30);                                  // type=2 ("image")
    v8i g1;
    g1[0] = (int)(2u << 16);      // workgroup_mask=0, data_size=2 (4B elems)
    g1[1] = (int)(256u << 16);    // tensor_dim0 = 256  (bits 79:48, low half)
    g1[2] = (int)(64u  << 16);    // tensor_dim1 = 64   (bits 111:80, low half)
    g1[3] = (int)(256u << 16);    // tile_dim0  = 256   (bits 127:112)
    g1[4] = (int)64;              // tile_dim1 = 64, tile_dim2 = 0
    g1[5] = (int)256;             // tensor_dim0_stride = 256 (low 32 of 48)
    g1[6] = 0;                    // stride0 hi / tensor_dim1_stride lo
    g1[7] = 0;                    // tensor_dim1_stride hi
    v4i gz4 = {0, 0, 0, 0};       // groups 2/3 unused for 2-D tensor
    v8i gz8 = {0, 0, 0, 0, 0, 0, 0, 0};
    __builtin_amdgcn_tensor_load_to_lds(g0, g1, gz4, gz4, gz8, 0);
    __builtin_amdgcn_s_wait_tensorcnt(0);
  }
  __syncthreads();

  // ---- Convert fp32 tile -> padded bf16 tile for WMMA fragment loads ----
  for (int i = tid; i < TS * H_DIM / 4; i += 256) {
    int row = i >> 6;             // 64 float4 per row
    int c4  = (i & 63) * 4;
    v4f v = *(const v4f*)&tile32[row * H_DIM + c4];
    v4bf t = { f2bf(v.x), f2bf(v.y), f2bf(v.z), f2bf(v.w) };
    *(v4bf*)&tile[row * LSTR + c4] = t;
  }
  __syncthreads();

  const int wave = tid >> 5, lane = tid & 31;
  const int m = wave & 3;        // M tile (16 s-rows)
  const int half = wave >> 2;    // o-range half (0..127 / 128..255)
  const int laneN = lane & 15, laneHi = lane >> 4;

  // Per-lane o-column constants (column is fixed per lane per N-tile).
  float qv[8], vv[8];
#pragma unroll
  for (int nt = 0; nt < 8; ++nt) {
    int o = half * 128 + nt * 16 + laneN;
    qv[nt] = qb[b * H_DIM + o];
    vv[nt] = Va_w[o];
  }

  v8f acc[8] = {};
  const __bf16* arow = &tile[(m * 16 + laneN) * LSTR];

#pragma unroll 1
  for (int kt = 0; kt < 8; ++kt) {
    // A fragment in contiguous k order (same order as B):
    // lanes 0-15: K = k0+{0..15}; lanes 16-31: K = k0+{16..31}
    const int h0 = kt * 32 + laneHi * 16;
    v8bf alo = *(const v8bf*)(arow + h0);
    v8bf ahi = *(const v8bf*)(arow + h0 + 8);
    v16bf a = __builtin_shufflevector(alo, ahi,
        0,1,2,3,4,5,6,7,8,9,10,11,12,13,14,15);
#pragma unroll
    for (int nt = 0; nt < 8; ++nt) {
      // B fragment: lane column o = n0+laneN, 16 contiguous h from Ua^T row o
      int o = half * 128 + nt * 16 + laneN;
      const __bf16* bp = ua + (size_t)o * H_DIM + kt * 32 + laneHi * 16;
      v16bf bm = *(const v16bf*)bp;
      acc[nt] = __builtin_amdgcn_wmma_f32_16x16x32_bf16(
          false, a, false, bm, (short)0, acc[nt], false, false);
    }
  }

  // Epilogue: tanh + Va dot, reduce across the 16 N-columns per half-wave.
#pragma unroll
  for (int r = 0; r < 8; ++r) {
    float p = 0.f;
#pragma unroll
    for (int nt = 0; nt < 8; ++nt)
      p += tanhf(acc[nt][r] + qv[nt]) * vv[nt];
    p += __shfl_xor(p, 1, 32);
    p += __shfl_xor(p, 2, 32);
    p += __shfl_xor(p, 4, 32);
    p += __shfl_xor(p, 8, 32);
    if (laneN == 0)  // C/D layout: row = r (lanes<16) or r+8 (lanes>=16)
      atomicAdd(&sc[m * 16 + r + laneHi * 8], p);  // 2 commutative addends
  }
  __syncthreads();
  if (tid < TS)
    scores[(size_t)b * S_LEN + s0 + tid] = sc[tid] + Va_b[0];
}

// ---- Phase B: per-batch softmax over S (deterministic tree reduce) ----
__global__ void __launch_bounds__(256)
softmax_kernel(float* __restrict__ scores) {
  const int b = blockIdx.x, tid = threadIdx.x;
  __shared__ float red[256];
  float* scb = scores + (size_t)b * S_LEN;
  float mx = -3.0e38f;
  for (int i = tid; i < S_LEN; i += 256) mx = fmaxf(mx, scb[i]);
  red[tid] = mx; __syncthreads();
  for (int st = 128; st > 0; st >>= 1) {
    if (tid < st) red[tid] = fmaxf(red[tid], red[tid + st]);
    __syncthreads();
  }
  mx = red[0]; __syncthreads();
  float sum = 0.f;
  for (int i = tid; i < S_LEN; i += 256) {
    float e = __expf(scb[i] - mx);
    scb[i] = e;
    sum += e;
  }
  red[tid] = sum; __syncthreads();
  for (int st = 128; st > 0; st >>= 1) {
    if (tid < st) red[tid] += red[tid + st];
    __syncthreads();
  }
  float inv = 1.f / red[0];
  for (int i = tid; i < S_LEN; i += 256) scb[i] *= inv;
}

// ---- Phase C: partial context per 256-row s-chunk ----
__global__ void __launch_bounds__(256)
ctx_partial_kernel(const float* __restrict__ keys, const float* __restrict__ w,
                   float* __restrict__ part) {
  const int chunk = blockIdx.x, b = blockIdx.y, h = threadIdx.x;
  __shared__ float wsh[256];
  const int s0 = chunk * 256;
  wsh[h] = w[(size_t)b * S_LEN + s0 + h];
  __syncthreads();
  const float* kb = keys + ((size_t)b * S_LEN + s0) * H_DIM;
  float a = 0.f;
  for (int j = 0; j < 256; ++j)
    a += wsh[j] * kb[(size_t)j * H_DIM + h];   // coalesced across h
  part[((size_t)b * 16 + chunk) * H_DIM + h] = a;
}

// ---- Phase D: deterministic reduction of 16 partials -> d_out ----
__global__ void __launch_bounds__(256)
ctx_reduce_kernel(const float* __restrict__ part, float* __restrict__ out) {
  const int b = blockIdx.x, h = threadIdx.x;
  float a = 0.f;
#pragma unroll
  for (int c = 0; c < 16; ++c)
    a += part[((size_t)b * 16 + c) * H_DIM + h];
  out[b * H_DIM + h] = a;
}

extern "C" void kernel_launch(void* const* d_in, const int* in_sizes, int n_in,
                              void* d_out, int out_size, void* d_ws, size_t ws_size,
                              hipStream_t stream) {
  const float* query = (const float*)d_in[0];
  const float* keys  = (const float*)d_in[1];
  const float* Wa_w  = (const float*)d_in[2];
  const float* Wa_b  = (const float*)d_in[3];
  const float* Ua_w  = (const float*)d_in[4];
  const float* Ua_b  = (const float*)d_in[5];
  const float* Va_w  = (const float*)d_in[6];
  const float* Va_b  = (const float*)d_in[7];
  float* out = (float*)d_out;

  char* ws = (char*)d_ws;
  float*  qb     = (float*)(ws);                               // 32 KB
  __bf16* ua     = (__bf16*)(ws + 32768);                      // 128 KB
  float*  scores = (float*)(ws + 32768 + 131072);              // 512 KB
  float*  part   = (float*)(ws + 32768 + 131072 + 524288);     // 128 KB

  qproj_kernel<<<B_N, H_DIM, 0, stream>>>(query, Wa_w, Wa_b, Ua_b, qb);
  cvt_ua_kernel<<<(H_DIM * H_DIM) / 256, 256, 0, stream>>>(Ua_w, ua);

  dim3 gA(S_LEN / TS, B_N);                                    // 64 x 32 blocks
  score_kernel<<<gA, 256, 0, stream>>>(keys, ua, qb, Va_w, Va_b, scores);

  softmax_kernel<<<B_N, 256, 0, stream>>>(scores);

  dim3 gC(S_LEN / 256, B_N);                                   // 16 x 32 blocks
  ctx_partial_kernel<<<gC, 256, 0, stream>>>(keys, scores, part);
  ctx_reduce_kernel<<<B_N, H_DIM, 0, stream>>>(part, out);
}
